// BasicBlock_67422396613116
// MI455X (gfx1250) — 8-bit matrix operand order repaired and checked by reading, compiled, not run on hardware
//
#include <hip/hip_runtime.h>

typedef __attribute__((ext_vector_type(8))) int v8i;

static constexpr int N_ = 32, C_ = 64, H_ = 128, W_ = 128;

__global__ __launch_bounds__(256) void pack_x_kernel(const float* __restrict__ x,
                                                     signed char* __restrict__ sx) {
  const size_t i = (size_t)blockIdx.x * 256 + threadIdx.x;
  const int g = (int)(i & 15);
  const int w = (int)((i >> 4) & 127);
  const int h = (int)((i >> 11) & 127);
  const int n = (int)(i >> 18);
  int out = 0;
#pragma unroll
  for (int j = 0; j < 4; ++j) {
    const int c = (g << 2) + j;
    const float v = x[((((size_t)n * C_ + c) * H_ + h) * W_) + w];
    const int s = (v > 0.f) - (v < 0.f);
    out |= (s & 0xff) << (8 * j);
  }
  *(int*)(sx + (i << 2)) = out;
}

__global__ void pack_w_kernel(const float* __restrict__ w, signed char* __restrict__ wp) {
  const int idx = blockIdx.x * 256 + threadIdx.x;
  if (idx >= 9 * 4 * 32 * 32) return;
  const int i    = idx & 3;
  const int d    = (idx >> 2) & 7;
  const int lane = (idx >> 5) & 31;
  const int ct   = (idx >> 10) & 3;
  const int t    = idx >> 12;
  const int cout = (ct << 4) + (lane & 15);
  const int hf   = lane >> 4;
  const int K    = ((d >> 2) << 5) + (hf << 4) + ((d & 3) << 2) + i;
  const int kh = t / 3, kw = t % 3;
  const float v = w[((((size_t)cout * C_) + K) * 3 + kh) * 3 + kw];
  wp[idx] = (signed char)((v > 0.f) - (v < 0.f));
}

__global__ __launch_bounds__(256) void bconv_wmma(const signed char* __restrict__ sx,
                                                  const signed char* __restrict__ wp,
                                                  short* __restrict__ y,
                                                  int* __restrict__ gsum,
                                                  unsigned long long* __restrict__ gsq) {
  __shared__ int4 tile4[4 * 130 * 4];
  __shared__ int s_sum[64];
  __shared__ int s_sq[64];

  const int tid = threadIdx.x;
  const int n  = blockIdx.x >> 6;
  const int hp = (blockIdx.x & 63) << 1;

  if (tid < 64) { s_sum[tid] = 0; s_sq[tid] = 0; }

  for (int idx = tid; idx < 4 * 130 * 4; idx += 256) {
    const int q   = idx & 3;
    const int pix = idx >> 2;
    const int r   = pix / 130;
    const int c   = pix - r * 130;
    const int gh  = hp + r - 1;
    const int gw  = c - 1;
    int4 v = make_int4(0, 0, 0, 0);
    if ((unsigned)gh < (unsigned)H_ && (unsigned)gw < (unsigned)W_)
      v = *(const int4*)(sx + ((((size_t)n * H_ + gh) * W_ + gw) << 6) + (q << 4));
    tile4[idx] = v;
  }
  __syncthreads();

  const int lane = tid & 31;
  const int wv   = tid >> 5;
  const int col  = lane & 15;
  const int hf   = lane >> 4;
  const signed char* tile = (const signed char*)tile4;

  v8i acc[2][4] = {};

#pragma unroll
  for (int t = 0; t < 9; ++t) {
    const int kh = t / 3, kw = t % 3;
    v8i B[2];
#pragma unroll
    for (int r = 0; r < 2; ++r) {
      const signed char* pb =
          tile + (((r + kh) * 130 + (wv << 4) + col + kw) << 6);
      const int4 b0 = *(const int4*)(pb + (hf << 4));
      const int4 b1 = *(const int4*)(pb + 32 + (hf << 4));
      B[r][0] = b0.x; B[r][1] = b0.y; B[r][2] = b0.z; B[r][3] = b0.w;
      B[r][4] = b1.x; B[r][5] = b1.y; B[r][6] = b1.z; B[r][7] = b1.w;
    }
#pragma unroll
    for (int ct = 0; ct < 4; ++ct) {
      const int4* pa = (const int4*)(wp + (size_t)((((t << 2) + ct) << 5) + lane) * 32);
      const int4 a0 = pa[0], a1 = pa[1];
      v8i A;
      A[0] = a0.x; A[1] = a0.y; A[2] = a0.z; A[3] = a0.w;
      A[4] = a1.x; A[5] = a1.y; A[6] = a1.z; A[7] = a1.w;
#pragma unroll
      for (int r = 0; r < 2; ++r) {
        acc[r][ct] = __builtin_amdgcn_wmma_i32_16x16x64_iu8(
             true, A,  true, B[r], acc[r][ct],
             false,  false);
      }
    }
  }

#pragma unroll
  for (int r2 = 0; r2 < 2; ++r2) {
    const size_t pixBase =
        (((size_t)n * H_ + hp + r2) * W_ + (wv << 4) + col) << 6;
#pragma unroll
    for (int ct = 0; ct < 4; ++ct) {
      short o[8];
#pragma unroll
      for (int r = 0; r < 8; ++r) o[r] = (short)acc[r2][ct][r];
      *(int4*)(y + pixBase + (ct << 4) + (hf << 3)) = *(const int4*)o;
    }
  }

#pragma unroll
  for (int ct = 0; ct < 4; ++ct) {
#pragma unroll
    for (int r = 0; r < 8; ++r) {
      const int v0 = acc[0][ct][r], v1 = acc[1][ct][r];
      int s = v0 + v1;
      int q = v0 * v0 + v1 * v1;
#pragma unroll
      for (int m = 1; m < 16; m <<= 1) {
        s += __shfl_xor(s, m, 32);
        q += __shfl_xor(q, m, 32);
      }
      if (col == 0) {
        const int ch = (ct << 4) + (hf << 3) + r;
        atomicAdd(&s_sum[ch], s);
        atomicAdd(&s_sq[ch], q);
      }
    }
  }
  __syncthreads();
  if (tid < 64) {
    atomicAdd(&gsum[tid], s_sum[tid]);
    atomicAdd(&gsq[tid], (unsigned long long)(unsigned int)s_sq[tid]);
  }
}

__global__ void bn_stats(const int* __restrict__ gsum,
                         const unsigned long long* __restrict__ gsq,
                         const float* __restrict__ gamma,
                         const float* __restrict__ beta,
                         float* __restrict__ ab) {
  const int c = threadIdx.x;
  if (c >= 64) return;
  const double inv  = 1.0 / (double)((size_t)N_ * H_ * W_);
  const double mean = (double)gsum[c] * inv;
  const double var  = (double)gsq[c] * inv - mean * mean;
  const float rstd  = (float)rsqrt(var + 1e-5);
  const float a     = gamma[c] * rstd;
  ab[c]      = a;
  ab[64 + c] = beta[c] - a * (float)mean;
}

__global__ __launch_bounds__(256) void bn_sign(const short* __restrict__ y,
                                               const float* __restrict__ ab,
                                               signed char* __restrict__ sxo) {
  const size_t i = (size_t)blockIdx.x * 256 + threadIdx.x;
  const int cg = (int)(i & 15) << 2;
  const short4 v = *(const short4*)(y + (i << 2));
  const short vv[4] = {v.x, v.y, v.z, v.w};
  int out = 0;
#pragma unroll
  for (int j = 0; j < 4; ++j) {
    const float t = ab[cg + j] * (float)vv[j] + ab[64 + cg + j];
    const int s = (t > 0.f) - (t < 0.f);
    out |= (s & 0xff) << (8 * j);
  }
  *(int*)(sxo + (i << 2)) = out;
}

__global__ __launch_bounds__(256) void bn_residual_out(const short* __restrict__ y,
                                                       const float* __restrict__ ab,
                                                       const float* __restrict__ x,
                                                       float* __restrict__ out) {
  const size_t i = (size_t)blockIdx.x * 256 + threadIdx.x;
  const int w = (int)(i & 127);
  const int h = (int)((i >> 7) & 127);
  const int c = (int)((i >> 14) & 63);
  const int n = (int)(i >> 20);
  const size_t yi = ((((size_t)n * H_ + h) * W_ + w) << 6) + c;
  const float v = ab[c] * (float)y[yi] + ab[64 + c] + x[i];
  out[i] = fminf(1.f, fmaxf(-1.f, v));
}

extern "C" void kernel_launch(void* const* d_in, const int* in_sizes, int n_in,
                              void* d_out, int out_size, void* d_ws, size_t ws_size,
                              hipStream_t stream) {
  (void)in_sizes; (void)n_in; (void)out_size; (void)ws_size;
  const float* x  = (const float*)d_in[0];
  const float* w1 = (const float*)d_in[1];
  const float* g1 = (const float*)d_in[2];
  const float* b1 = (const float*)d_in[3];
  const float* w2 = (const float*)d_in[4];
  const float* g2 = (const float*)d_in[5];
  const float* b2 = (const float*)d_in[6];
  float* out = (float*)d_out;

  char* ws = (char*)d_ws;
  int*                sum1 = (int*)(ws + 0);
  unsigned long long* sq1  = (unsigned long long*)(ws + 256);
  int*                sum2 = (int*)(ws + 768);
  unsigned long long* sq2  = (unsigned long long*)(ws + 1024);
  float*              ab1  = (float*)(ws + 1536);
  float*              ab2  = (float*)(ws + 2048);
  signed char*        wp1  = (signed char*)(ws + 4096);
  signed char*        wp2  = (signed char*)(ws + 4096 + 36864);
  signed char*        sx1  = (signed char*)(ws + 131072);
  signed char*        sx2  = sx1 + (size_t)33554432;
  short*              y1   = (short*)(sx2 + (size_t)33554432);
  short*              y2   = y1 + (size_t)33554432;

  hipMemsetAsync(ws, 0, 1536, stream);

  pack_x_kernel<<<32768, 256, 0, stream>>>(x, sx1);
  pack_w_kernel<<<144, 256, 0, stream>>>(w1, wp1);
  pack_w_kernel<<<144, 256, 0, stream>>>(w2, wp2);

  bconv_wmma<<<N_ * H_ / 2, 256, 0, stream>>>(sx1, wp1, y1, sum1, sq1);
  bn_stats<<<1, 64, 0, stream>>>(sum1, sq1, g1, b1, ab1);
  bn_sign<<<32768, 256, 0, stream>>>(y1, ab1, sx2);

  bconv_wmma<<<N_ * H_ / 2, 256, 0, stream>>>(sx2, wp2, y2, sum2, sq2);
  bn_stats<<<1, 64, 0, stream>>>(sum2, sq2, g2, b2, ab2);
  bn_residual_out<<<131072, 256, 0, stream>>>(y2, ab2, x, out);
}
